// GINo_32658931319022
// MI455X (gfx1250) — compile-verified, matrix operand order checked by reading, not run on hardware
//
#include <hip/hip_runtime.h>
#include <hip/hip_bf16.h>
#include <math.h>

// ---------------------------------------------------------------------------
// GIN(3 layers) + BiLSTM JumpingKnowledge + attention + FC tail for MI455X.
// Dense GEMMs: v_wmma_f32_16x16x32_f16, one A-fragment reused across 4 B
// column tiles per wave (4 WMMAs per A-load pair). Edge aggregation: float4
// loads + float atomics (agg buffer is <=51MB -> L2 resident at 192MB).
// ---------------------------------------------------------------------------

typedef _Float16 f16;
typedef _Float16 v16h __attribute__((ext_vector_type(16)));
typedef _Float16 v8h  __attribute__((ext_vector_type(8)));
typedef _Float16 v4h  __attribute__((ext_vector_type(4)));
typedef float    v8f  __attribute__((ext_vector_type(8)));

#define GSTRIDE(i, n) for (long i = (long)blockIdx.x * blockDim.x + threadIdx.x; \
                           i < (n); i += (long)gridDim.x * blockDim.x)

// ---------------- elementwise / utility kernels (16B vectorized) -----------

__global__ void k_cast16v(const float4* __restrict__ s, v4h* __restrict__ d, long n4) {
  GSTRIDE(i, n4) {
    float4 v = s[i];
    v4h o; o.x = (f16)v.x; o.y = (f16)v.y; o.z = (f16)v.z; o.w = (f16)v.w;
    d[i] = o;
  }
}

__global__ void k_zero32v(float4* __restrict__ d, long n4) {
  const float4 z = make_float4(0.f, 0.f, 0.f, 0.f);
  GSTRIDE(i, n4) d[i] = z;
}

__global__ void k_zero16v(v4h* __restrict__ d, long n4) {
  v4h z; z.x = (f16)0.f; z.y = (f16)0.f; z.z = (f16)0.f; z.w = (f16)0.f;
  GSTRIDE(i, n4) d[i] = z;
}

__global__ void k_addvec(const float* __restrict__ a, const float* __restrict__ b,
                         float* __restrict__ d, int n) {
  int i = blockIdx.x * blockDim.x + threadIdx.x;
  if (i < n) d[i] = a[i] + b[i];
}

// segment_sum(h[src], dst): one (edge, 4-feature group) per thread.
// gshift = log2(F/4); float4 feature load, 4 float atomics into L2.
__global__ void k_scatter(const float* __restrict__ h, const int* __restrict__ src,
                          const int* __restrict__ dst, float* __restrict__ agg,
                          long E, int gshift) {
  const int gmask = (1 << gshift) - 1;
  const long total = E << gshift;
  GSTRIDE(i, total) {
    long e = i >> gshift;
    int  g = (int)(i & gmask);
    const float4 hv = *(const float4*)(h + (((long)src[e]) << (gshift + 2)) + g * 4);
    float* ap = agg + (((long)dst[e]) << (gshift + 2)) + g * 4;
    atomicAdd(ap + 0, hv.x);
    atomicAdd(ap + 1, hv.y);
    atomicAdd(ap + 2, hv.z);
    atomicAdd(ap + 3, hv.w);
  }
}

// z = h + agg, cast to f16 for the WMMA GEMM that follows.
__global__ void k_addcast(const float4* __restrict__ h, const float4* __restrict__ agg,
                          v4h* __restrict__ z16, long n4) {
  GSTRIDE(i, n4) {
    float4 a = h[i], b = agg[i];
    v4h o;
    o.x = (f16)(a.x + b.x); o.y = (f16)(a.y + b.y);
    o.z = (f16)(a.z + b.z); o.w = (f16)(a.w + b.w);
    z16[i] = o;
  }
}

// ---------------- WMMA GEMM:  out = act(A[M x K] @ W[NOUT x K]^T + bias) ----
// A row-major (K contiguous), W row-major [NOUT x K] (== B column-major):
// both fragments are K-contiguous 16B loads per lane:
//   lane L (L<16):  row/col = L,     K = {0..7},{16..23}  (+kk)
//   lane L (L>=16): row/col = L-16,  K = {8..15},{24..31} (+kk)
// C/D: acc[r] -> row = row0 + 8*(lane>>4) + r, col = col0 + (lane&15).
// Each wave owns 16 rows x 64 cols (4 col tiles): the A fragment is reused by
// 4 WMMAs. Block = 4 waves = 64 rows x 64 cols. M%16==0; tail waves guarded.
template <int K, int NOUT, int RELU, bool BIAS, bool CIN, bool O32, bool O16>
__global__ __launch_bounds__(128) void k_gemm(const f16* __restrict__ A,
                                              const f16* __restrict__ W,
                                              const float* __restrict__ bias,
                                              const float* __restrict__ Cin,
                                              float* __restrict__ Of32,
                                              f16* __restrict__ Of16, int M) {
  const int lane = threadIdx.x & 31;
  const int wave = threadIdx.x >> 5;
  const int row0 = (blockIdx.x * 4 + wave) * 16;
  if (row0 >= M) return;                       // M%16==0 -> live tiles are full
  const int col0 = blockIdx.y * 64;
  const int t    = lane & 15;                  // row (A) / col (B) within tile
  const int hi   = lane >> 4;                  // K sub-offset 0/8, C row offset 0/8

  const f16* __restrict__ arow  = A + (long)(row0 + t) * K + hi * 8;
  const f16* __restrict__ wrow0 = W + (long)(col0 + t) * K + hi * 8;

  v8f acc[4];
  const v8f zero = {};
#pragma unroll
  for (int j = 0; j < 4; ++j) {
    if (CIN) {
#pragma unroll
      for (int r = 0; r < 8; ++r)
        acc[j][r] = Cin[(long)(row0 + hi * 8 + r) * NOUT + col0 + j * 16 + t];
    } else {
      acc[j] = zero;
    }
  }

#pragma unroll
  for (int kk = 0; kk < K; kk += 32) {
    v8h a0 = *(const v8h*)(arow + kk);
    v8h a1 = *(const v8h*)(arow + kk + 16);
    v16h a = __builtin_shufflevector(a0, a1, 0,1,2,3,4,5,6,7,8,9,10,11,12,13,14,15);
#pragma unroll
    for (int j = 0; j < 4; ++j) {
      const f16* wr = wrow0 + (long)j * 16 * K;
      v8h b0 = *(const v8h*)(wr + kk);
      v8h b1 = *(const v8h*)(wr + kk + 16);
      v16h b = __builtin_shufflevector(b0, b1, 0,1,2,3,4,5,6,7,8,9,10,11,12,13,14,15);
      acc[j] = __builtin_amdgcn_wmma_f32_16x16x32_f16(false, a, false, b,
                                                      (short)0, acc[j], false, false);
    }
  }

#pragma unroll
  for (int j = 0; j < 4; ++j) {
    const float badd = BIAS ? bias[col0 + j * 16 + t] : 0.0f;
#pragma unroll
    for (int r = 0; r < 8; ++r) {
      float v = acc[j][r] + badd;
      if (RELU) v = v > 0.0f ? v : 0.0f;
      const long idx = (long)(row0 + hi * 8 + r) * NOUT + col0 + j * 16 + t;
      if (O32) Of32[idx] = v;
      if (O16) Of16[idx] = (f16)v;
    }
  }
}

// ---------------- LSTM cell (gate order i,f,g,o; H=96, gates=384) ----------
__device__ __forceinline__ float sigm_(float x) { return 1.0f / (1.0f + __expf(-x)); }

__global__ void k_lstm_cell(const float* __restrict__ g, float* __restrict__ c,
                            f16* __restrict__ h16, long Nn) {
  const long total = Nn * 96;
  GSTRIDE(i, total) {
    long n = i / 96;
    int  j = (int)(i % 96);
    const float* gr = g + n * 384;
    float cn = sigm_(gr[96 + j]) * c[i] + sigm_(gr[j]) * tanhf(gr[192 + j]);
    c[i] = cn;
    h16[i] = (f16)(sigm_(gr[288 + j]) * tanhf(cn));
  }
}

// alpha logit contribution: dot(h[n,:96], Watt_seg[:96]) per node.
__global__ void k_alpha(const f16* __restrict__ h16, const float* __restrict__ wseg,
                        float* __restrict__ alpha, long Nn) {
  GSTRIDE(n, Nn) {
    const f16* hr = h16 + n * 96;
    float s = 0.0f;
#pragma unroll
    for (int j = 0; j < 96; ++j) s += (float)hr[j] * wseg[j];
    alpha[n] = s;
  }
}

// softmax over L=3 layers + weighted combine of seq -> xjk (f16 for Wlin GEMM)
__global__ void k_jk(const float* __restrict__ seq, const float* __restrict__ aF,
                     const float* __restrict__ aB, const float* __restrict__ battp,
                     f16* __restrict__ xjk16, long Nn) {
  const float bat = battp[0];
  GSTRIDE(n, Nn) {
    float a0 = aF[n] + aB[n] + bat;
    float a1 = aF[Nn + n] + aB[Nn + n] + bat;
    float a2 = aF[2 * Nn + n] + aB[2 * Nn + n] + bat;
    float mx = fmaxf(a0, fmaxf(a1, a2));
    float e0 = __expf(a0 - mx), e1 = __expf(a1 - mx), e2 = __expf(a2 - mx);
    float inv = 1.0f / (e0 + e1 + e2);
    e0 *= inv; e1 *= inv; e2 *= inv;
    const float* s0 = seq + n * 64;
    const float* s1 = seq + (Nn + n) * 64;
    const float* s2 = seq + (2 * Nn + n) * 64;
    f16* o = xjk16 + n * 64;
#pragma unroll
    for (int c = 0; c < 64; ++c) o[c] = (f16)(s0[c] * e0 + s1[c] * e1 + s2[c] * e2);
  }
}

// v[n] = LeakyReLU(xo[n,:] . Wfc1 + bfc1)
__global__ void k_fc1(const float* __restrict__ xo, const float* __restrict__ Wfc1,
                      const float* __restrict__ bfc1, float* __restrict__ v, long Nn) {
  GSTRIDE(n, Nn) {
    float s = bfc1[0];
    const float* r = xo + n * 64;
#pragma unroll
    for (int c = 0; c < 64; ++c) s += r[c] * Wfc1[c];
    v[n] = s >= 0.0f ? s : 0.01f * s;
  }
}

// final scalar: deterministic 2-stage reduction of Wfc2 . v + bfc2
__global__ void k_dotp(const float* __restrict__ Wfc2, const float* __restrict__ v,
                       float* __restrict__ parts, long Nn) {
  __shared__ float red[256];
  float s = 0.0f;
  GSTRIDE(i, Nn) s += Wfc2[i] * v[i];
  red[threadIdx.x] = s;
  __syncthreads();
  for (int off = 128; off > 0; off >>= 1) {
    if (threadIdx.x < off) red[threadIdx.x] += red[threadIdx.x + off];
    __syncthreads();
  }
  if (threadIdx.x == 0) parts[blockIdx.x] = red[0];
}

__global__ void k_dotf(const float* __restrict__ parts, int np,
                       const float* __restrict__ bfc2, float* __restrict__ out) {
  if (threadIdx.x == 0 && blockIdx.x == 0) {
    float s = bfc2[0];
    for (int i = 0; i < np; ++i) s += parts[i];
    out[0] = s;
  }
}

// ---------------------------------------------------------------------------

static inline int nblk(long n, int t) {
  long b = (n + t - 1) / t;
  if (b > (1L << 20)) b = 1L << 20;
  if (b < 1) b = 1;
  return (int)b;
}

extern "C" void kernel_launch(void* const* d_in, const int* in_sizes, int n_in,
                              void* d_out, int out_size, void* d_ws, size_t ws_size,
                              hipStream_t stream) {
  // ---- inputs (setup_inputs dict order) ----
  const float* x    = (const float*)d_in[0];
  const int*   ei   = (const int*)d_in[1];
  const float* Wa[3] = {(const float*)d_in[2], (const float*)d_in[6],  (const float*)d_in[10]};
  const float* ba[3] = {(const float*)d_in[3], (const float*)d_in[7],  (const float*)d_in[11]};
  const float* Wb[3] = {(const float*)d_in[4], (const float*)d_in[8],  (const float*)d_in[12]};
  const float* bb[3] = {(const float*)d_in[5], (const float*)d_in[9],  (const float*)d_in[13]};
  const float* Wih_f = (const float*)d_in[14];
  const float* Whh_f = (const float*)d_in[15];
  const float* bih_f = (const float*)d_in[16];
  const float* bhh_f = (const float*)d_in[17];
  const float* Wih_b = (const float*)d_in[18];
  const float* Whh_b = (const float*)d_in[19];
  const float* bih_b = (const float*)d_in[20];
  const float* bhh_b = (const float*)d_in[21];
  const float* Watt  = (const float*)d_in[22];
  const float* batt  = (const float*)d_in[23];
  const float* Wlin  = (const float*)d_in[24];
  const float* blin  = (const float*)d_in[25];
  const float* Wfc1  = (const float*)d_in[26];
  const float* bfc1  = (const float*)d_in[27];
  const float* Wfc2  = (const float*)d_in[28];
  const float* bfc2  = (const float*)d_in[29];

  const long Nn = in_sizes[28];          // Wfc2 is (1, N); N=100000, mult of 16
  const long E  = in_sizes[1] / 2;
  const int* src = ei;
  const int* dst = ei + E;

  // ---- workspace carve (256B aligned) ----
  size_t cur = 0;
  auto carve = [&](size_t bytes) -> char* {
    char* p = (char*)d_ws + cur;
    cur += (bytes + 255) & ~(size_t)255;
    return p;
  };
  float* seq   = (float*)carve((size_t)3 * Nn * 64 * 4);   // layer outputs (f32)
  f16*   seq16 = (f16*)  carve((size_t)3 * Nn * 64 * 2);   // f16 copy for GEMMs
  float* alF   = (float*)carve((size_t)3 * Nn * 4);        // fwd attention logits
  float* alB   = (float*)carve((size_t)3 * Nn * 4);        // bwd attention logits
  // big scratch, reused per phase; LSTM phase is largest: g + h16 + c
  char* scr = carve((size_t)Nn * 384 * 4 + (size_t)Nn * 96 * 2 + (size_t)Nn * 96 * 4);
  // f16 weights
  f16* wA16[3]; f16* wB16[3];
  wA16[0] = (f16*)carve((size_t)64 * 128 * 2);
  for (int l = 1; l < 3; ++l) wA16[l] = (f16*)carve((size_t)64 * 64 * 2);
  for (int l = 0; l < 3; ++l) wB16[l] = (f16*)carve((size_t)64 * 64 * 2);
  f16* wihF16 = (f16*)carve((size_t)384 * 64 * 2);
  f16* whhF16 = (f16*)carve((size_t)384 * 96 * 2);
  f16* wihB16 = (f16*)carve((size_t)384 * 64 * 2);
  f16* whhB16 = (f16*)carve((size_t)384 * 96 * 2);
  f16* wlin16 = (f16*)carve((size_t)64 * 64 * 2);
  float* biasF = (float*)carve(384 * 4);
  float* biasB = (float*)carve(384 * 4);
  float* parts = (float*)carve(1024 * 4);
  (void)ws_size; (void)n_in; (void)out_size;

  const int T = 256;
  const int Mi = (int)Nn;

  // ---- weight preparation (f32 -> f16, vectorized) ----
  auto cast16 = [&](const float* s, f16* d, long n) {
    k_cast16v<<<nblk(n / 4, T), T, 0, stream>>>((const float4*)s, (v4h*)d, n / 4);
  };
  cast16(Wa[0], wA16[0], 64 * 128);
  cast16(Wa[1], wA16[1], 64 * 64);
  cast16(Wa[2], wA16[2], 64 * 64);
  for (int l = 0; l < 3; ++l) cast16(Wb[l], wB16[l], 64 * 64);
  cast16(Wih_f, wihF16, 384 * 64);
  cast16(Whh_f, whhF16, 384 * 96);
  cast16(Wih_b, wihB16, 384 * 64);
  cast16(Whh_b, whhB16, 384 * 96);
  cast16(Wlin,  wlin16, 64 * 64);
  k_addvec<<<2, T, 0, stream>>>(bih_f, bhh_f, biasF, 384);
  k_addvec<<<2, T, 0, stream>>>(bih_b, bhh_b, biasB, 384);

  const dim3 gemmBlk(128);
  const unsigned gx = (unsigned)((Nn + 63) / 64);   // 4 row-tiles per block
  const dim3 grid64(gx, 1);
  const dim3 grid384(gx, 6);

  // ---- GIN layers ----
  {
    float* agg   = (float*)scr;
    f16*   z16   = (f16*)(scr + (size_t)Nn * 128 * 4);
    f16*   mid16 = (f16*)(scr + (size_t)Nn * 128 * 4 + (size_t)Nn * 128 * 2);

    const float* hin = x;
    // layer 0: K=128 (gshift = log2(128/4) = 5)
    k_zero32v<<<nblk(Nn * 32, T), T, 0, stream>>>((float4*)agg, Nn * 32);
    k_scatter<<<nblk(E * 32, T), T, 0, stream>>>(hin, src, dst, agg, E, 5);
    k_addcast<<<nblk(Nn * 32, T), T, 0, stream>>>((const float4*)hin, (const float4*)agg,
                                                  (v4h*)z16, Nn * 32);
    k_gemm<128, 64, 1, true, false, false, true>
        <<<grid64, gemmBlk, 0, stream>>>(z16, wA16[0], ba[0], nullptr, nullptr, mid16, Mi);
    k_gemm<64, 64, 1, true, false, true, true>
        <<<grid64, gemmBlk, 0, stream>>>(mid16, wB16[0], bb[0], nullptr, seq, seq16, Mi);
    // layers 1,2: K=64 (gshift = 4)
    for (int l = 1; l < 3; ++l) {
      hin = seq + (size_t)(l - 1) * Nn * 64;
      k_zero32v<<<nblk(Nn * 16, T), T, 0, stream>>>((float4*)agg, Nn * 16);
      k_scatter<<<nblk(E * 16, T), T, 0, stream>>>(hin, src, dst, agg, E, 4);
      k_addcast<<<nblk(Nn * 16, T), T, 0, stream>>>((const float4*)hin, (const float4*)agg,
                                                    (v4h*)z16, Nn * 16);
      k_gemm<64, 64, 1, true, false, false, true>
          <<<grid64, gemmBlk, 0, stream>>>(z16, wA16[l], ba[l], nullptr, nullptr, mid16, Mi);
      k_gemm<64, 64, 1, true, false, true, true>
          <<<grid64, gemmBlk, 0, stream>>>(mid16, wB16[l], bb[l], nullptr,
                                           seq + (size_t)l * Nn * 64,
                                           seq16 + (size_t)l * Nn * 64, Mi);
    }
  }

  // ---- BiLSTM JumpingKnowledge (L=3, H=96, gates=384) ----
  {
    float* g    = (float*)scr;
    f16*   h16c = (f16*)(scr + (size_t)Nn * 384 * 4);
    float* ccar = (float*)(scr + (size_t)Nn * 384 * 4 + (size_t)Nn * 96 * 2);

    for (int dir = 0; dir < 2; ++dir) {
      const f16*   wih = dir ? wihB16 : wihF16;
      const f16*   whh = dir ? whhB16 : whhF16;
      const float* bias = dir ? biasB : biasF;
      const float* wseg = dir ? (Watt + 96) : Watt;
      float*       al   = dir ? alB : alF;

      k_zero16v<<<nblk(Nn * 24, T), T, 0, stream>>>((v4h*)h16c, Nn * 24);
      k_zero32v<<<nblk(Nn * 24, T), T, 0, stream>>>((float4*)ccar, Nn * 24);
      for (int s = 0; s < 3; ++s) {
        const int l = dir ? (2 - s) : s;
        k_gemm<64, 384, 0, true, false, true, false>
            <<<grid384, gemmBlk, 0, stream>>>(seq16 + (size_t)l * Nn * 64, wih, bias,
                                              nullptr, g, nullptr, Mi);
        k_gemm<96, 384, 0, false, true, true, false>
            <<<grid384, gemmBlk, 0, stream>>>(h16c, whh, nullptr, g, g, nullptr, Mi);
        k_lstm_cell<<<nblk(Nn * 96, T), T, 0, stream>>>(g, ccar, h16c, Nn);
        k_alpha<<<nblk(Nn, T), T, 0, stream>>>(h16c, wseg, al + (size_t)l * Nn, Nn);
      }
    }
  }

  // ---- attention combine + FC tail ----
  {
    f16*   xjk16 = (f16*)scr;
    float* xo    = (float*)(scr + (size_t)Nn * 64 * 2);
    float* vv    = (float*)(scr + (size_t)Nn * 64 * 2 + (size_t)Nn * 64 * 4);

    k_jk<<<nblk(Nn, T), T, 0, stream>>>(seq, alF, alB, batt, xjk16, Nn);
    k_gemm<64, 64, 0, true, false, true, false>
        <<<grid64, gemmBlk, 0, stream>>>(xjk16, wlin16, blin, nullptr, xo, nullptr, Mi);
    k_fc1<<<nblk(Nn, T), T, 0, stream>>>(xo, Wfc1, bfc1, vv, Nn);
    k_dotp<<<1024, 256, 0, stream>>>(Wfc2, vv, parts, Nn);
    k_dotf<<<1, 64, 0, stream>>>(parts, 1024, bfc2, (float*)d_out);
  }
}
